// CANet_34411277975758
// MI455X (gfx1250) — matrix operand order repaired and checked by reading, not compiled, not run on hardware
//
#include <hip/hip_runtime.h>
#include <hip/hip_bf16.h>
#include <math.h>

#define C_INCH   12
#define C_NOISE  3
#define HID      128
#define PERC     48
#define KIM      108          // im2col depth: 12 ch * 3 * 3
#define TDIM     256
#define NB       16
#define NH       256
#define NW       256
#define HW       (NH*NW)

#define PIX_PER_BLOCK 128
#define THREADS       256

// single LDS activation buffer stride (bf16 elems): 128 used + 8 pad (16B)
// keeps every b128 access 16B aligned (272 = 17*16) and staggers banks.
#define AH_S 136

typedef __attribute__((ext_vector_type(16))) __bf16 v16bf;
typedef __attribute__((ext_vector_type(8)))  float  v8f;

union Frag {
    v16bf v;
    uint4 q[2];
};

// workspace layout (bytes)
#define WS_EMB_OFF 0                               // 48 floats
#define WS_B1_OFF  256                             // 128 floats: b1' = b1 + W1*bp
#define WS_W1_OFF  (WS_B1_OFF + HID*4)             // 128*128 bf16: W1' (K 108 -> pad 128)
#define WS_W2_OFF  (WS_W1_OFF + HID*HID*2)         // 128*128 bf16
#define WS_W3_OFF  (WS_W2_OFF + HID*HID*2)         // 16*128  bf16 (N pad 12 -> 16)

// ---------------------------------------------------------------------------
// Prep: time embedding, conv-folded first-layer weights, bf16 conversions.
// ---------------------------------------------------------------------------
__global__ __launch_bounds__(256) void canet_prep(
    const int* __restrict__ t,  const float* __restrict__ wp,
    const float* __restrict__ bp, const float* __restrict__ w1,
    const float* __restrict__ b1, const float* __restrict__ w2,
    const float* __restrict__ w3, const float* __restrict__ w_time,
    const float* __restrict__ b_time, unsigned char* __restrict__ ws)
{
    const int tid = threadIdx.x;
    float*  emb = (float*)(ws + WS_EMB_OFF);
    float*  b1p = (float*)(ws + WS_B1_OFF);
    __bf16* w1p = (__bf16*)(ws + WS_W1_OFF);
    __bf16* w2b = (__bf16*)(ws + WS_W2_OFF);
    __bf16* w3b = (__bf16*)(ws + WS_W3_OFF);

    // time embedding: silu(posenc(t)) @ w_time.T + b_time   (16 x 3)
    if (tid < NB * C_NOISE) {
        int b = tid / C_NOISE, c = tid % C_NOISE;
        float tf  = (float)t[b];
        float acc = b_time[c];
        for (int j = 0; j < TDIM / 2; ++j) {
            float invf = powf(10000.0f, -(float)j / 128.0f);
            float ang  = tf * invf;
            float ps   = sinf(ang);
            float pc   = cosf(ang);
            float ss   = ps / (1.0f + expf(-ps));
            float sc   = pc / (1.0f + expf(-pc));
            acc += ss * w_time[c*TDIM + j] + sc * w_time[c*TDIM + 128 + j];
        }
        emb[b*C_NOISE + c] = acc;
    }

    // b1' = b1 + W1 @ b_perceive
    if (tid < HID) {
        float acc = b1[tid];
        for (int oc = 0; oc < PERC; ++oc) acc += w1[tid*PERC + oc] * bp[oc];
        b1p[tid] = acc;
    }

    // W1' = W1 @ W_dw_expanded : [128][108], zero-padded K to 128
    for (int i = tid; i < HID*HID; i += 256) {
        int n = i >> 7, k = i & 127;
        float val = 0.0f;
        if (k < KIM) {
            int ic = k / 9, j = k % 9;
            #pragma unroll
            for (int m = 0; m < 4; ++m) {
                int oc = ic*4 + m;
                val += w1[n*PERC + oc] * wp[oc*9 + j];
            }
        }
        w1p[i] = (__bf16)val;
    }

    for (int i = tid; i < HID*HID; i += 256) w2b[i] = (__bf16)w2[i];

    for (int i = tid; i < 16*HID; i += 256) {
        int n = i >> 7, k = i & 127;
        w3b[i] = (__bf16)((n < C_INCH) ? w3[n*HID + k] : 0.0f);
    }
}

// ---------------------------------------------------------------------------
// Main: im2col staging + 3 WMMA GEMM layers, single in-place LDS buffer.
// Block = 256 threads = 8 waves; each wave owns a 16-pixel M-tile.
// B fragments are double-buffered so each WMMA consumes a fragment whose
// load was issued one WMMA earlier (overlaps L0 latency with matrix pipe).
// ---------------------------------------------------------------------------
__global__ __launch_bounds__(256, 2) void canet_main(
    const float* __restrict__ x, const float* __restrict__ hidden,
    const float* __restrict__ b2, const unsigned char* __restrict__ ws,
    float* __restrict__ out)
{
    __shared__ __bf16 act[PIX_PER_BLOCK * AH_S];   // im2col -> h1 -> h2 (in place)

    const float*  emb = (const float*)(ws + WS_EMB_OFF);
    const float*  b1p = (const float*)(ws + WS_B1_OFF);
    const __bf16* w1p = (const __bf16*)(ws + WS_W1_OFF);
    const __bf16* w2b = (const __bf16*)(ws + WS_W2_OFF);
    const __bf16* w3b = (const __bf16*)(ws + WS_W3_OFF);

    const int tid = threadIdx.x;
    const int blk = blockIdx.x;
    const int b   = blk >> 9;          // 512 blocks per image
    const int rr0 = blk & 511;
    const int row = rr0 >> 1;          // image row
    const int w0  = (rr0 & 1) << 7;    // column base (0 or 128)

    // ---------------- im2col staging (no FMAs, loads + bf16 stores) --------
    {
        const int p   = tid & 127;         // pixel within block
        const int icb = (tid >> 7) * 6;    // this thread covers 6 input channels
        const int wc  = w0 + p;            // image column
        if (tid < PIX_PER_BLOCK) {         // zero the K pad (108..127)
            #pragma unroll
            for (int k = KIM; k < HID; ++k) act[tid*AH_S + k] = (__bf16)0.0f;
        }
        #pragma unroll
        for (int i = 0; i < 6; ++i) {
            const int ic = icb + i;
            const float* src = (ic < C_NOISE)
                ? x      + ((size_t)b*C_NOISE          + ic)           * HW
                : hidden + ((size_t)b*(C_INCH-C_NOISE) + (ic-C_NOISE)) * HW;
            #pragma unroll
            for (int kh = 0; kh < 3; ++kh) {
                const int ih  = row + kh - 1;
                const bool oh = (unsigned)ih < (unsigned)NH;
                #pragma unroll
                for (int kw = 0; kw < 3; ++kw) {
                    const int iw = wc + kw - 1;
                    const bool ok = oh && ((unsigned)iw < (unsigned)NW);
                    float v = ok ? src[ih*NW + iw] : 0.0f;
                    act[p*AH_S + ic*9 + kh*3 + kw] = (__bf16)v;
                }
            }
        }
    }
    __syncthreads();

    // ---------------- WMMA lane geometry ----------------
    const int lane  = tid & 31;
    const int wv    = tid >> 5;
    const int wbase = wv * 16;             // wave's first pixel row
    const int lm    = lane & 15;           // M row (A) / N column (B,C)
    const int hi    = lane >> 4;           // lane-half selector
    const int rowA  = wbase + lm;
    const int kbA   = hi * 16;             // A-fragment K base (contiguous: k = 16*hi + s)

    const __bf16* wRow1 = w1p + lm*HID + hi*16;   // per-lane base into W1'
    const __bf16* wRow2 = w2b + lm*HID + hi*16;   // per-lane base into W2
    const __bf16* wRow3 = w3b + lm*HID + hi*16;   // per-lane base into W3

    // ---------------- layer 1' : [16x128] x [128x128] (conv folded in) ----
    v8f c1[8] = {};
    #pragma unroll
    for (int s = 0; s < 4; ++s) {
        Frag a;
        const __bf16* ap = &act[rowA*AH_S + s*32 + kbA];
        a.q[0] = *(const uint4*)ap;          // K kb..kb+7
        a.q[1] = *(const uint4*)(ap + 8);    // K kb+8..kb+15
        Frag bfr[2];
        bfr[0].q[0] = *(const uint4*)(wRow1 + s*32);
        bfr[0].q[1] = *(const uint4*)(wRow1 + s*32 + 8);
        #pragma unroll
        for (int n = 0; n < 8; ++n) {
            if (n < 7) {   // prefetch fragment n+1 before consuming n
                const __bf16* bpp = wRow1 + (n+1)*16*HID + s*32;
                bfr[(n+1)&1].q[0] = *(const uint4*)bpp;
                bfr[(n+1)&1].q[1] = *(const uint4*)(bpp + 8);
            }
            c1[n] = __builtin_amdgcn_wmma_f32_16x16x32_bf16(
                        false, a.v, false, bfr[n&1].v, (short)0, c1[n], false, false);
        }
    }
    #pragma unroll
    for (int n = 0; n < 8; ++n) {
        const float bias = b1p[n*16 + lm];
        #pragma unroll
        for (int r = 0; r < 8; ++r) {
            float v = c1[n][r] + bias;
            v = v > 0.0f ? v : 0.0f;                       // relu
            act[(wbase + r + hi*8)*AH_S + n*16 + lm] = (__bf16)v;
        }
    }
    __syncthreads();

    // ---------------- layer 2 : [16x128] x [128x128] (in place) -----------
    v8f c2[8] = {};
    #pragma unroll
    for (int s = 0; s < 4; ++s) {
        Frag a;
        const __bf16* ap = &act[rowA*AH_S + s*32 + kbA];
        a.q[0] = *(const uint4*)ap;
        a.q[1] = *(const uint4*)(ap + 8);
        Frag bfr[2];
        bfr[0].q[0] = *(const uint4*)(wRow2 + s*32);
        bfr[0].q[1] = *(const uint4*)(wRow2 + s*32 + 8);
        #pragma unroll
        for (int n = 0; n < 8; ++n) {
            if (n < 7) {
                const __bf16* bpp = wRow2 + (n+1)*16*HID + s*32;
                bfr[(n+1)&1].q[0] = *(const uint4*)bpp;
                bfr[(n+1)&1].q[1] = *(const uint4*)(bpp + 8);
            }
            c2[n] = __builtin_amdgcn_wmma_f32_16x16x32_bf16(
                        false, a.v, false, bfr[n&1].v, (short)0, c2[n], false, false);
        }
    }
    #pragma unroll
    for (int n = 0; n < 8; ++n) {
        const float bias = b2[n*16 + lm];
        #pragma unroll
        for (int r = 0; r < 8; ++r) {
            float v = c2[n][r] + bias;
            v = v > 0.0f ? v : 0.0f;
            act[(wbase + r + hi*8)*AH_S + n*16 + lm] = (__bf16)v;
        }
    }
    __syncthreads();

    // ---------------- layer 3 : [16x128] x [128x16] ----------------
    // double-buffer across k-steps (B varies with s here)
    v8f c3 = {};
    {
        Frag bfr[2];
        bfr[0].q[0] = *(const uint4*)(wRow3);
        bfr[0].q[1] = *(const uint4*)(wRow3 + 8);
        #pragma unroll
        for (int s = 0; s < 4; ++s) {
            if (s < 3) {
                bfr[(s+1)&1].q[0] = *(const uint4*)(wRow3 + (s+1)*32);
                bfr[(s+1)&1].q[1] = *(const uint4*)(wRow3 + (s+1)*32 + 8);
            }
            Frag a;
            const __bf16* ap = &act[rowA*AH_S + s*32 + kbA];
            a.q[0] = *(const uint4*)ap;
            a.q[1] = *(const uint4*)(ap + 8);
            c3 = __builtin_amdgcn_wmma_f32_16x16x32_bf16(
                     false, a.v, false, bfr[s&1].v, (short)0, c3, false, false);
        }
    }

    // ---------------- scatter outputs ----------------
    const int oc = lm;                 // output channel for this lane
    if (oc < C_INCH) {
        const bool  isNoise = (oc < C_NOISE);
        const float addv = isNoise ? emb[b*C_NOISE + oc] : 0.0f;
        const size_t noiseBase = (size_t)NB * (C_INCH - C_NOISE) * HW;
        #pragma unroll
        for (int r = 0; r < 8; ++r) {
            const int p  = wbase + r + hi*8;
            const int hw = row*NW + w0 + p;
            const float v = c3[r] + addv;
            size_t off = isNoise
                ? noiseBase + ((size_t)b*C_NOISE + oc)*HW + hw
                : ((size_t)b*(C_INCH - C_NOISE) + (oc - C_NOISE))*HW + hw;
            out[off] = v;
        }
    }
}

// ---------------------------------------------------------------------------
extern "C" void kernel_launch(void* const* d_in, const int* in_sizes, int n_in,
                              void* d_out, int out_size, void* d_ws, size_t ws_size,
                              hipStream_t stream)
{
    const float* x   = (const float*)d_in[0];
    const float* hid = (const float*)d_in[1];
    const int*   t   = (const int*)  d_in[2];
    const float* wp  = (const float*)d_in[3];
    const float* bp  = (const float*)d_in[4];
    const float* w1  = (const float*)d_in[5];
    const float* b1  = (const float*)d_in[6];
    const float* w2  = (const float*)d_in[7];
    const float* b2  = (const float*)d_in[8];
    const float* w3  = (const float*)d_in[9];
    const float* wt  = (const float*)d_in[10];
    const float* bt  = (const float*)d_in[11];
    float* out = (float*)d_out;
    unsigned char* ws = (unsigned char*)d_ws;

    canet_prep<<<dim3(1), dim3(256), 0, stream>>>(t, wp, bp, w1, b1, w2, w3, wt, bt, ws);

    const int nblocks = (NB * NH * NW) / PIX_PER_BLOCK;   // 8192
    canet_main<<<dim3(nblocks), dim3(THREADS), 0, stream>>>(
        x, hid, b2, (const unsigned char*)ws, out);
}
